// MultiHeadAttention_57853209477815
// MI455X (gfx1250) — matrix operand order repaired and checked by reading, not compiled, not run on hardware
//
#include <hip/hip_runtime.h>
#include <hip/hip_bf16.h>

typedef __attribute__((ext_vector_type(16))) __bf16 v16bf;
typedef __attribute__((ext_vector_type(8)))  float  v8f;

union BF16x16 { v16bf v; unsigned int u[8]; };
union BFPair  { __bf16 h[2]; unsigned int u; };

#define HH 768
#define LDDW 384   // 768 bf16 = 384 dwords per row

__device__ __forceinline__ v8f wmma_bf16(v16bf a, v16bf b, v8f c) {
    return __builtin_amdgcn_wmma_f32_16x16x32_bf16(false, a, false, b, (short)0, c, false, false);
}

__device__ __forceinline__ float phi_elu1(float x) {
    // elu(x)+1 : x>0 -> x+1 ; else exp(x)
    return x > 0.0f ? x + 1.0f : __expf(x);
}

// A fragment 16x32 (MxK), 16-bit: lanes 0-15 M=0..15 / K 0..15 ; lanes 16-31 K 16..31
__device__ __forceinline__ v16bf load_a_frag(const unsigned int* base, int row0, int k0, int lane) {
    int half = lane >> 4, m = lane & 15;
    BF16x16 f;
#pragma unroll
    for (int r = 0; r < 8; ++r) {
        int kk = k0 + half * 16 + 2 * r;
        f.u[r] = base[(row0 + m) * LDDW + (kk >> 1)];
    }
    return f.v;
}

// B fragment 32x16 (KxN): lanes 0-15 hold K=0..15 (N=lane), lanes 16-31 hold K=16..31
// B[k][n] = W[col0+n][kbase+k]  (pairs along k contiguous in W's row)
__device__ __forceinline__ v16bf load_b_frag(const unsigned int* base, int col0, int k0, int lane) {
    int half = lane >> 4, n = lane & 15;
    BF16x16 f;
#pragma unroll
    for (int r = 0; r < 8; ++r) {
        int kk = k0 + half * 16 + 2 * r;
        f.u[r] = base[(col0 + n) * LDDW + (kk >> 1)];
    }
    return f.v;
}

// ---------------- weight fp32 -> bf16 cast ----------------
__global__ __launch_bounds__(256) void wcast_kernel(const float* __restrict__ w,
                                                    __bf16* __restrict__ o, int n) {
    int i = blockIdx.x * 256 + threadIdx.x;
    if (i < n) o[i] = (__bf16)w[i];
}

// ---------------- LayerNorm(768) + bf16 cast ----------------
__global__ __launch_bounds__(256) void ln_kernel(const float* __restrict__ x,
                                                 const float* __restrict__ g,
                                                 const float* __restrict__ bta,
                                                 __bf16* __restrict__ y) {
    int row = blockIdx.x;
    int t = threadIdx.x;
    const float* xr = x + (size_t)row * HH;
    float v0 = xr[t], v1 = xr[t + 256], v2 = xr[t + 512];
    float s1 = v0 + v1 + v2;
    float s2 = v0 * v0 + v1 * v1 + v2 * v2;
#pragma unroll
    for (int off = 16; off >= 1; off >>= 1) {
        s1 += __shfl_down(s1, off, 32);
        s2 += __shfl_down(s2, off, 32);
    }
    __shared__ float ls1[8], ls2[8];
    int lane = t & 31, wv = t >> 5;
    if (lane == 0) { ls1[wv] = s1; ls2[wv] = s2; }
    __syncthreads();
    float t1 = 0.f, t2 = 0.f;
#pragma unroll
    for (int i = 0; i < 8; ++i) { t1 += ls1[i]; t2 += ls2[i]; }
    float mu = t1 * (1.0f / HH);
    float var = t2 * (1.0f / HH) - mu * mu;
    float rs = rsqrtf(var + 1e-5f);
    __bf16* yr = y + (size_t)row * HH;
    yr[t]       = (__bf16)((v0 - mu) * rs * g[t]       + bta[t]);
    yr[t + 256] = (__bf16)((v1 - mu) * rs * g[t + 256] + bta[t + 256]);
    yr[t + 512] = (__bf16)((v2 - mu) * rs * g[t + 512] + bta[t + 512]);
}

// ---------------- WMMA GEMM: out[M,768] = A[M,768](bf16) @ W(bf16)[768,768]^T + bias ----------------
// mode 0: identity -> bf16 out        (V projection)
// mode 1: phi=elu+1 -> bf16 out       (Q projection)
// mode 2: phi*mask  -> bf16 out       (K projection)
// mode 3: + residual -> fp32 out      (final FC)
__global__ __launch_bounds__(256) void gemm768_kernel(
    const unsigned int* __restrict__ A, const unsigned int* __restrict__ W,
    const float* __restrict__ bias, const unsigned char* __restrict__ mask,
    const float* __restrict__ residual, __bf16* __restrict__ outB,
    float* __restrict__ outF, int mode) {
    int lane = threadIdx.x & 31;
    int wave = threadIdx.x >> 5;
    int wy = wave >> 1, wx = wave & 1;
    int rbase = blockIdx.x * 256 + wy * 64;
    int cbase = blockIdx.y * 128 + wx * 64;

    v8f acc[4][4];
#pragma unroll
    for (int i = 0; i < 4; ++i)
#pragma unroll
        for (int j = 0; j < 4; ++j) acc[i][j] = (v8f){};

    for (int k0 = 0; k0 < HH; k0 += 32) {
        v16bf a[4], b[4];
#pragma unroll
        for (int mi = 0; mi < 4; ++mi) a[mi] = load_a_frag(A, rbase + mi * 16, k0, lane);
#pragma unroll
        for (int ni = 0; ni < 4; ++ni) b[ni] = load_b_frag(W, cbase + ni * 16, k0, lane);
#pragma unroll
        for (int mi = 0; mi < 4; ++mi)
#pragma unroll
            for (int ni = 0; ni < 4; ++ni) acc[mi][ni] = wmma_bf16(a[mi], b[ni], acc[mi][ni]);
    }

    int half = lane >> 4, n = lane & 15;
#pragma unroll
    for (int mi = 0; mi < 4; ++mi) {
#pragma unroll
        for (int r = 0; r < 8; ++r) {
            int row = rbase + mi * 16 + r + half * 8;
            float mk = (mode == 2) ? (float)mask[row] : 1.0f;
#pragma unroll
            for (int ni = 0; ni < 4; ++ni) {
                int col = cbase + ni * 16 + n;
                float v = acc[mi][ni][r] + bias[col];
                if (mode == 1) v = phi_elu1(v);
                else if (mode == 2) v = phi_elu1(v) * mk;
                size_t o = (size_t)row * HH + col;
                if (mode == 3) outF[o] = v + residual[o];
                else outB[o] = (__bf16)v;
            }
        }
    }
}

// ---------------- KV[b,h,m,d] = sum_s V[b,s,h,m] * K[b,s,h,d]  (one block per (b,h)) ----------------
__global__ __launch_bounds__(128) void kv_kernel(const unsigned int* __restrict__ Kb,
                                                 const unsigned int* __restrict__ Vb,
                                                 __bf16* __restrict__ KV, int S, int NH) {
    int bh = blockIdx.x;
    int b = bh / NH, h = bh % NH;
    __shared__ unsigned int KsU[1024];  // 32 s-rows x 64 bf16
    __shared__ unsigned int VsU[1024];
    __bf16* Ks = (__bf16*)KsU;
    __bf16* Vs = (__bf16*)VsU;
    int lane = threadIdx.x & 31;
    int wave = threadIdx.x >> 5;
    int half = lane >> 4, mm = lane & 15;

    v8f acc[4];
#pragma unroll
    for (int j = 0; j < 4; ++j) acc[j] = (v8f){};

    for (int s0 = 0; s0 < S; s0 += 32) {
#pragma unroll
        for (int i = 0; i < 8; ++i) {
            int idx = threadIdx.x + i * 128;      // dword 0..1023
            int s = idx >> 5, d2 = idx & 31;
            int g = (b * S + s0 + s) * LDDW + h * 32;
            KsU[idx] = Kb[g + d2];
            VsU[idx] = Vb[g + d2];
        }
        __syncthreads();
        // A[m][s] = Vs[s][m], strip rows m = wave*16 + mm
        BF16x16 fa;
#pragma unroll
        for (int r = 0; r < 8; ++r) {
            int kk = half * 16 + 2 * r;
            BFPair p;
            p.h[0] = Vs[kk * 64 + wave * 16 + mm];
            p.h[1] = Vs[(kk + 1) * 64 + wave * 16 + mm];
            fa.u[r] = p.u;
        }
#pragma unroll
        for (int j = 0; j < 4; ++j) {
            BF16x16 fb;  // B[s][d] = Ks[s][d]
#pragma unroll
            for (int r = 0; r < 8; ++r) {
                int kk = half * 16 + 2 * r;
                BFPair p;
                p.h[0] = Ks[kk * 64 + j * 16 + mm];
                p.h[1] = Ks[(kk + 1) * 64 + j * 16 + mm];
                fb.u[r] = p.u;
            }
            acc[j] = wmma_bf16(fa.v, fb.v, acc[j]);
        }
        __syncthreads();
    }
#pragma unroll
    for (int j = 0; j < 4; ++j)
#pragma unroll
        for (int r = 0; r < 8; ++r)
            KV[((size_t)bh * 64 + wave * 16 + r + half * 8) * 64 + j * 16 + mm] =
                (__bf16)acc[j][r];
}

// ---------------- Ksum[b,h,d] = sum_s K[b,s,h,d] ----------------
__global__ __launch_bounds__(256) void ksum_kernel(const __bf16* __restrict__ Kb,
                                                   float* __restrict__ Ksum, int S, int NH) {
    int bh = blockIdx.x;
    int b = bh / NH, h = bh % NH;
    int d = threadIdx.x & 63, srow = threadIdx.x >> 6;
    float acc = 0.f;
    for (int s = srow; s < S; s += 4)
        acc += (float)Kb[((size_t)(b * S + s)) * HH + h * 64 + d];
    __shared__ float red[256];
    red[threadIdx.x] = acc;
    __syncthreads();
    if (srow == 0)
        Ksum[bh * 64 + d] = red[d] + red[64 + d] + red[128 + d] + red[192 + d];
}

// ---------------- Z[b,s,h] = 1 / (Q . Ksum + eps) ----------------
__global__ __launch_bounds__(256) void z_kernel(const __bf16* __restrict__ Qb,
                                                const float* __restrict__ Ksum,
                                                float* __restrict__ Z, int total, int S, int NH) {
    int idx = blockIdx.x * 256 + threadIdx.x;
    if (idx >= total) return;
    int h = idx % NH;
    int bs = idx / NH;      // b*S + s
    int b = bs / S;
    const __bf16* q = Qb + (size_t)bs * HH + h * 64;
    const float* ks = Ksum + (b * NH + h) * 64;
    float dot = 0.f;
#pragma unroll
    for (int d = 0; d < 64; ++d) dot += (float)q[d] * ks[d];
    Z[idx] = 1.0f / (dot + 1e-6f);
}

// ---------------- attn[b,s,h,m] = Z * sum_d Q[b,s,h,d]*KV[b,h,m,d] ----------------
__global__ __launch_bounds__(256) void attn_out_kernel(const unsigned int* __restrict__ Qb,
                                                       const unsigned int* __restrict__ KVb,
                                                       const float* __restrict__ Z,
                                                       __bf16* __restrict__ attn, int S, int NH) {
    int bh = blockIdx.x;
    int b = bh / NH, h = bh % NH;
    int wave = threadIdx.x >> 5, lane = threadIdx.x & 31;
    int half = lane >> 4;
    int sTile = blockIdx.y * 128 + wave * 16;

    v8f acc[4];
#pragma unroll
    for (int j = 0; j < 4; ++j) acc[j] = (v8f){};

#pragma unroll
    for (int kc = 0; kc < 64; kc += 32) {
        BF16x16 fa;  // A[srow][d] = Q[b, sTile+srow, h, kc+..]
        int m = lane & 15;
#pragma unroll
        for (int r = 0; r < 8; ++r) {
            int kk = kc + half * 16 + 2 * r;
            fa.u[r] = Qb[(size_t)(b * S + sTile + m) * LDDW + ((h * 64 + kk) >> 1)];
        }
#pragma unroll
        for (int j = 0; j < 4; ++j) {
            BF16x16 fb;  // B[k=d][n=m] = KV[b,h, j*16+n, d]  (pairs along d contiguous)
#pragma unroll
            for (int r = 0; r < 8; ++r) {
                int kk = kc + half * 16 + 2 * r;
                fb.u[r] = KVb[(size_t)(bh * 64 + j * 16 + (lane & 15)) * 32 + (kk >> 1)];
            }
            acc[j] = wmma_bf16(fa.v, fb.v, acc[j]);
        }
    }
    int n = lane & 15;
#pragma unroll
    for (int r = 0; r < 8; ++r) {
        int s = sTile + r + half * 8;
        float z = Z[(size_t)(b * S + s) * NH + h];
#pragma unroll
        for (int j = 0; j < 4; ++j)
            attn[(size_t)(b * S + s) * HH + h * 64 + j * 16 + n] = (__bf16)(acc[j][r] * z);
    }
}

extern "C" void kernel_launch(void* const* d_in, const int* in_sizes, int n_in,
                              void* d_out, int out_size, void* d_ws, size_t ws_size,
                              hipStream_t stream) {
    const int B = 4, S = 4096, NH = 12;
    const int M = B * S;  // 16384 rows

    const float* q    = (const float*)d_in[0];
    const float* k    = (const float*)d_in[1];
    const float* v    = (const float*)d_in[2];
    const unsigned char* mask = (const unsigned char*)d_in[3];  // jnp bool -> 1 byte
    const float* q_g = (const float*)d_in[4];
    const float* q_b = (const float*)d_in[5];
    const float* k_g = (const float*)d_in[6];
    const float* k_b = (const float*)d_in[7];
    const float* v_g = (const float*)d_in[8];
    const float* v_b = (const float*)d_in[9];
    const float* Wq  = (const float*)d_in[10];
    const float* bq  = (const float*)d_in[11];
    const float* Wk  = (const float*)d_in[12];
    const float* bk  = (const float*)d_in[13];
    const float* Wv  = (const float*)d_in[14];
    const float* bv  = (const float*)d_in[15];
    const float* Wfc = (const float*)d_in[16];
    const float* bfc = (const float*)d_in[17];

    // ----- workspace layout -----
    char* ws = (char*)d_ws;
    const size_t WSZ = (size_t)HH * HH * 2;     // 1,179,648 B per bf16 weight
    const size_t XSZ = (size_t)M * HH * 2;      // 25,165,824 B per bf16 activation
    __bf16* Wqb  = (__bf16*)(ws + 0 * WSZ);
    __bf16* Wkb  = (__bf16*)(ws + 1 * WSZ);
    __bf16* Wvb  = (__bf16*)(ws + 2 * WSZ);
    __bf16* Wfcb = (__bf16*)(ws + 3 * WSZ);
    char* base = ws + 4 * WSZ;
    __bf16* Xnq = (__bf16*)(base + 0 * XSZ);
    __bf16* Xnk = (__bf16*)(base + 1 * XSZ);
    __bf16* Xnv = (__bf16*)(base + 2 * XSZ);
    __bf16* Qb  = (__bf16*)(base + 3 * XSZ);
    __bf16* Kb  = (__bf16*)(base + 4 * XSZ);
    __bf16* Vb  = (__bf16*)(base + 5 * XSZ);
    // reuse: attn overlays Xnq (dead after proj GEMMs); KV/Ksum/Z overlay Xnk
    __bf16* attn = Xnq;
    __bf16* KVb  = Xnk;                                    // 393,216 B
    float*  Ksum = (float*)((char*)Xnk + 524288);          // 12,288 B
    float*  Zbuf = (float*)((char*)Xnk + 1048576);         // 786,432 B

    // 1) weights -> bf16
    {
        int n = HH * HH;
        dim3 g((n + 255) / 256);
        wcast_kernel<<<g, 256, 0, stream>>>(Wq,  Wqb,  n);
        wcast_kernel<<<g, 256, 0, stream>>>(Wk,  Wkb,  n);
        wcast_kernel<<<g, 256, 0, stream>>>(Wv,  Wvb,  n);
        wcast_kernel<<<g, 256, 0, stream>>>(Wfc, Wfcb, n);
    }
    // 2) LayerNorm + cast
    ln_kernel<<<M, 256, 0, stream>>>(q, q_g, q_b, Xnq);
    ln_kernel<<<M, 256, 0, stream>>>(k, k_g, k_b, Xnk);
    ln_kernel<<<M, 256, 0, stream>>>(v, v_g, v_b, Xnv);

    // 3) projections (WMMA): Q = phi(LN(q)Wq^T+bq), K = phi(.)*mask, V = identity
    dim3 gg(M / 256, HH / 128);
    gemm768_kernel<<<gg, 256, 0, stream>>>((const unsigned int*)Xnq, (const unsigned int*)Wqb,
                                           bq, nullptr, nullptr, Qb, nullptr, 1);
    gemm768_kernel<<<gg, 256, 0, stream>>>((const unsigned int*)Xnk, (const unsigned int*)Wkb,
                                           bk, mask, nullptr, Kb, nullptr, 2);
    gemm768_kernel<<<gg, 256, 0, stream>>>((const unsigned int*)Xnv, (const unsigned int*)Wvb,
                                           bv, nullptr, nullptr, Vb, nullptr, 0);

    // 4) KV and Ksum
    kv_kernel<<<B * NH, 128, 0, stream>>>((const unsigned int*)Kb, (const unsigned int*)Vb,
                                          KVb, S, NH);
    ksum_kernel<<<B * NH, 256, 0, stream>>>(Kb, Ksum, S, NH);

    // 5) Z normalizer
    {
        int total = M * NH;
        z_kernel<<<(total + 255) / 256, 256, 0, stream>>>(Qb, Ksum, Zbuf, total, S, NH);
    }

    // 6) attention output (WMMA)
    attn_out_kernel<<<dim3(B * NH, S / 128), 256, 0, stream>>>(
        (const unsigned int*)Qb, (const unsigned int*)KVb, Zbuf, attn, S, NH);

    // 7) final FC + residual -> fp32 out
    gemm768_kernel<<<gg, 256, 0, stream>>>((const unsigned int*)attn, (const unsigned int*)Wfcb,
                                           bfc, nullptr, q, nullptr, (float*)d_out, 3);
}
